// GIN_GNN_3865470566845
// MI455X (gfx1250) — compile-verified, matrix operand order checked by reading, not run on hardware
//
#include <hip/hip_runtime.h>
#include <hip/hip_bf16.h>

#define NNODE  100000
#define NEDGE  1600000
#define HDIM   256
#define NGRAPH 64

typedef __attribute__((ext_vector_type(16))) __bf16 v16bf;
typedef __attribute__((ext_vector_type(8)))  float  v8f;
typedef __attribute__((ext_vector_type(4)))  float  f32x4;

union FragU { v16bf v; f32x4 f[2]; };

// Load one 16x32 bf16 WMMA operand fragment (A rows or Wt rows == B cols).
// Per CDNA5 ISA 16-bit A layout: lanes<16 hold K 0..7 & 16..23, lanes>=16 hold
// K 8..15 & 24..31 -> two b128 loads at +hi*8 elems and +hi*8+16 elems.
static __device__ __forceinline__ v16bf load_frag16(const __bf16* __restrict__ base, int hi) {
  FragU u;
  const f32x4* p = reinterpret_cast<const f32x4*>(base + (hi << 3));
  u.f[0] = p[0];
  u.f[1] = p[2];
  return u.v;
}

static __device__ __forceinline__ float silu_f(float y) {
  return y / (1.0f + __expf(-y));
}

static __device__ __forceinline__ void atomicMaxF(float* addr, float v) {
  unsigned int* ua = reinterpret_cast<unsigned int*>(addr);
  unsigned int old = *ua;
  while (__uint_as_float(old) < v) {
    unsigned int assumed = old;
    old = atomicCAS(ua, assumed, __float_as_uint(v));
    if (old == assumed) break;
  }
}

// ---------------------------------------------------------------------------
// Fused bf16 WMMA GEMM: out = epilogue(A[M,K] @ Wt[NOUT,K]^T + bias)
// Block: 256 threads = 8 waves, each wave owns a NOUT/8-column slice and BOTH
// 16-row groups of the 32-row block tile (B fragments reused across 2 WMMAs,
// zero cross-wave B redundancy -> ~1.5 b128 loads per WMMA at NOUT=512).
// EPI 0: bias only                       -> outF (fp32)
// EPI 1: bias + LayerNorm + SiLU         -> outB (bf16)
// EPI 2: bias + LN + SiLU + a*h+(1-a)*x  -> outF (fp32, overwrites x)
// ---------------------------------------------------------------------------
template<int NOUT, int EPI>
__global__ __launch_bounds__(256) void gemm_wmma(
    const __bf16* __restrict__ A, const __bf16* __restrict__ Wt,
    const float* __restrict__ bias, int K,
    const float* __restrict__ lng, const float* __restrict__ lnb,
    const float* __restrict__ alphaP, const float* __restrict__ xoldF,
    float* __restrict__ outF, __bf16* __restrict__ outB)
{
  constexpr int NT = NOUT / 128;          // WMMA n-tiles per wave (8 n-waves)
  const int lane = threadIdx.x & 31;
  const int wave = threadIdx.x >> 5;      // 0..7, pure n-split
  const int hi   = lane >> 4;
  const int ln15 = lane & 15;
  const int noff = wave * (NOUT / 8);
  const size_t mrow = (size_t)blockIdx.x * 32;

  v8f acc[2][NT];
  #pragma unroll
  for (int mg = 0; mg < 2; ++mg)
    #pragma unroll
    for (int t = 0; t < NT; ++t)
      #pragma unroll
      for (int r = 0; r < 8; ++r) acc[mg][t][r] = 0.0f;

  const __bf16* aBase0 = A + (mrow + (size_t)ln15) * (size_t)K;
  const __bf16* aBase1 = A + (mrow + 16 + (size_t)ln15) * (size_t)K;
  for (int kk = 0; kk < K; kk += 32) {
    v16bf a0 = load_frag16(aBase0 + kk, hi);
    v16bf a1 = load_frag16(aBase1 + kk, hi);
    #pragma unroll
    for (int t = 0; t < NT; ++t) {
      int col = noff + t * 16 + ln15;
      v16bf bf = load_frag16(Wt + (size_t)col * K + kk, hi);
      acc[0][t] = __builtin_amdgcn_wmma_f32_16x16x32_bf16(
          false, a0, false, bf, (short)0, acc[0][t], false, false);
      acc[1][t] = __builtin_amdgcn_wmma_f32_16x16x32_bf16(
          false, a1, false, bf, (short)0, acc[1][t], false, false);
    }
  }

  if constexpr (EPI == 0) {
    #pragma unroll
    for (int mg = 0; mg < 2; ++mg) {
      size_t rbase = mrow + mg * 16 + (size_t)hi * 8;
      #pragma unroll
      for (int t = 0; t < NT; ++t) {
        int col = noff + t * 16 + ln15;
        float b = bias[col];
        #pragma unroll
        for (int r = 0; r < 8; ++r) {
          float v = acc[mg][t][r] + b;
          size_t idx = (rbase + r) * (size_t)NOUT + col;
          outF[idx] = v;
          if (outB) outB[idx] = (__bf16)v;
        }
      }
    }
  } else {
    __shared__ float redS[32][8];
    __shared__ float redQ[32][8];
    float s[2][8], q[2][8];
    #pragma unroll
    for (int mg = 0; mg < 2; ++mg)
      #pragma unroll
      for (int r = 0; r < 8; ++r) { s[mg][r] = 0.f; q[mg][r] = 0.f; }
    #pragma unroll
    for (int mg = 0; mg < 2; ++mg)
      #pragma unroll
      for (int t = 0; t < NT; ++t) {
        int col = noff + t * 16 + ln15;
        float b = bias[col];
        #pragma unroll
        for (int r = 0; r < 8; ++r) {
          float v = acc[mg][t][r] + b;
          acc[mg][t][r] = v;
          s[mg][r] += v; q[mg][r] += v * v;
        }
      }
    // butterfly all-reduce across the 16 lanes of each half-wave (rows differ by hi)
    #pragma unroll
    for (int m = 1; m < 16; m <<= 1) {
      #pragma unroll
      for (int mg = 0; mg < 2; ++mg)
        #pragma unroll
        for (int r = 0; r < 8; ++r) {
          s[mg][r] += __shfl_xor(s[mg][r], m, 32);
          q[mg][r] += __shfl_xor(q[mg][r], m, 32);
        }
    }
    if (ln15 == 0) {
      #pragma unroll
      for (int mg = 0; mg < 2; ++mg)
        #pragma unroll
        for (int r = 0; r < 8; ++r) {
          int rl = mg * 16 + hi * 8 + r;
          redS[rl][wave] = s[mg][r]; redQ[rl][wave] = q[mg][r];
        }
    }
    __syncthreads();
    float mean[2][8], rstd[2][8];
    #pragma unroll
    for (int mg = 0; mg < 2; ++mg)
      #pragma unroll
      for (int r = 0; r < 8; ++r) {
        int rl = mg * 16 + hi * 8 + r;
        float S = 0.f, Q = 0.f;
        #pragma unroll
        for (int w = 0; w < 8; ++w) { S += redS[rl][w]; Q += redQ[rl][w]; }
        float mu = S * (1.0f / NOUT);
        mean[mg][r] = mu;
        rstd[mg][r] = rsqrtf(fmaxf(Q * (1.0f / NOUT) - mu * mu, 0.0f) + 1e-5f);
      }
    float alpha = (EPI == 2) ? *alphaP : 0.0f;
    #pragma unroll
    for (int mg = 0; mg < 2; ++mg) {
      size_t rbase = mrow + mg * 16 + (size_t)hi * 8;
      #pragma unroll
      for (int t = 0; t < NT; ++t) {
        int col = noff + t * 16 + ln15;
        float gg = lng[col], bb = lnb[col];
        #pragma unroll
        for (int r = 0; r < 8; ++r) {
          float y  = (acc[mg][t][r] - mean[mg][r]) * rstd[mg][r] * gg + bb;
          float sv = silu_f(y);
          size_t idx = (rbase + r) * (size_t)NOUT + col;
          if constexpr (EPI == 1) {
            outB[idx] = (__bf16)sv;
          } else {
            float xn = alpha * sv + (1.0f - alpha) * xoldF[idx];
            outF[idx] = xn;
          }
        }
      }
    }
  }
}

// W[K,NOUT] fp32 -> Wt[NOUT,K] bf16
__global__ void convert_transpose(const float* __restrict__ W, __bf16* __restrict__ Wt,
                                  int K, int NOUT) {
  int tid = blockIdx.x * blockDim.x + threadIdx.x;
  if (tid >= K * NOUT) return;
  int n = tid / K, k = tid % K;
  Wt[(size_t)n * K + k] = (__bf16)W[(size_t)k * NOUT + n];
}

// Per-node embeddings + pos-MLP -> concat feature [N,448] bf16
__global__ __launch_bounds__(128) void node_features(
    const int* __restrict__ atoms, const int* __restrict__ tags,
    const float* __restrict__ pos,
    const float* __restrict__ atom_emb, const float* __restrict__ tag_emb,
    const float* __restrict__ pw1, const float* __restrict__ pb1,
    const float* __restrict__ pw2, const float* __restrict__ pb2,
    __bf16* __restrict__ feat)
{
  __shared__ float t[128];
  int n = blockIdx.x;
  int j = threadIdx.x;
  float p0 = pos[n * 3 + 0], p1 = pos[n * 3 + 1], p2 = pos[n * 3 + 2];
  float a = p0 * pw1[0 * 128 + j] + p1 * pw1[1 * 128 + j] + p2 * pw1[2 * 128 + j] + pb1[j];
  t[j] = silu_f(a);
  __syncthreads();
  float acc = pb2[j];
  #pragma unroll 4
  for (int k = 0; k < 128; ++k) acc += t[k] * pw2[k * 128 + j];
  size_t base = (size_t)n * 448;
  int atom = atoms[n];
  feat[base + j]         = (__bf16)atom_emb[(size_t)atom * 256 + j];
  feat[base + 128 + j]   = (__bf16)atom_emb[(size_t)atom * 256 + 128 + j];
  if (j < 64) feat[base + 256 + j] = (__bf16)tag_emb[(size_t)tags[n] * 64 + j];
  feat[base + 320 + j]   = (__bf16)acc;
}

// agg[dst] += x[src]  (L2-resident atomics; one thread = one edge x 4 columns)
__global__ void scatter_add(const float* __restrict__ xf, const int* __restrict__ ei,
                            float* __restrict__ agg) {
  long long tid = (long long)blockIdx.x * blockDim.x + threadIdx.x;
  int e = (int)(tid >> 6);
  int q = (int)(tid & 63);
  if (e >= NEDGE) return;
  int s = ei[e], d = ei[NEDGE + e];
  f32x4 v = reinterpret_cast<const f32x4*>(xf)[(size_t)s * 64 + q];
  float* dst = agg + (size_t)d * 256 + q * 4;
  atomicAdd(dst + 0, v.x);
  atomicAdd(dst + 1, v.y);
  atomicAdd(dst + 2, v.z);
  atomicAdd(dst + 3, v.w);
}

// xa = bf16((1+eps)*x + agg)
__global__ void combine_gin(const float* __restrict__ xf, const float* __restrict__ agg,
                            const float* __restrict__ epsP, __bf16* __restrict__ xa) {
  long long tid = (long long)blockIdx.x * blockDim.x + threadIdx.x;
  if (tid >= (long long)NNODE * HDIM) return;
  float e = *epsP;
  xa[tid] = (__bf16)((1.0f + e) * xf[tid] + agg[tid]);
}

// pooled[g][j] += x[n] @ pw[:,j] + pb[j]
__global__ __launch_bounds__(256) void pool_proj(
    const float* __restrict__ xf, const float* __restrict__ pw,
    const float* __restrict__ pb, const int* __restrict__ batch,
    float* __restrict__ pooled)
{
  int j = threadIdx.x & 63;
  int n = blockIdx.x * 4 + (threadIdx.x >> 6);
  const float* xr = xf + (size_t)n * 256;
  float acc = pb[j];
  #pragma unroll 4
  for (int c = 0; c < 256; ++c) acc += xr[c] * pw[c * 64 + j];
  atomicAdd(&pooled[(size_t)batch[n] * 64 + j], acc);
}

// segment add / max / counts over final x
__global__ void final_pool(const float* __restrict__ xf, const int* __restrict__ batch,
                           float* __restrict__ addb, float* __restrict__ maxb,
                           float* __restrict__ counts) {
  long long tid = (long long)blockIdx.x * blockDim.x + threadIdx.x;
  if (tid >= (long long)NNODE * HDIM) return;
  int n = (int)(tid >> 8);
  int c = (int)(tid & 255);
  int g = batch[n];
  float v = xf[tid];
  atomicAdd(&addb[(size_t)g * 256 + c], v);
  atomicMaxF(&maxb[(size_t)g * 256 + c], v);
  if (c == 0) atomicAdd(&counts[g], 1.0f);
}

__global__ void fill_neginf(float* p, int n) {
  int tid = blockIdx.x * blockDim.x + threadIdx.x;
  if (tid < n) p[tid] = -__builtin_inff();
}

// comb[g] = [mean(256), max(256), add(256), pooled0/safe, pooled1/safe, pooled2/safe]
__global__ void build_comb(const float* __restrict__ addb, const float* __restrict__ maxb,
                           const float* __restrict__ counts, const float* __restrict__ pooled,
                           float* __restrict__ comb) {
  int tid = blockIdx.x * blockDim.x + threadIdx.x;
  if (tid >= NGRAPH * 960) return;
  int g = tid / 960, c = tid % 960;
  float cnt = counts[g];
  float safe = fmaxf(cnt, 1.0f);
  float v;
  if (c < 256)       v = addb[g * 256 + c] / safe;
  else if (c < 512)  { float m = maxb[g * 256 + (c - 256)]; v = (cnt > 0.0f) ? m : 0.0f; }
  else if (c < 768)  v = addb[g * 256 + (c - 512)];
  else {
    int p = (c - 768) / 64, jj = (c - 768) % 64;
    v = pooled[(size_t)p * NGRAPH * 64 + g * 64 + jj] / safe;
  }
  comb[tid] = v;
}

static __device__ __forceinline__ float blockSum(float v, float* red) {
  int j = threadIdx.x;
  red[j] = v;
  __syncthreads();
  for (int s = 128; s > 0; s >>= 1) {
    if (j < s) red[j] += red[j + s];
    __syncthreads();
  }
  float r = red[0];
  __syncthreads();
  return r;
}

// Final MLP readout: one block per graph.
__global__ __launch_bounds__(256) void readout(
    const float* __restrict__ comb,
    const float* __restrict__ w1, const float* __restrict__ b1,
    const float* __restrict__ g1, const float* __restrict__ e1,
    const float* __restrict__ w2, const float* __restrict__ b2,
    const float* __restrict__ g2, const float* __restrict__ e2,
    const float* __restrict__ w3, const float* __restrict__ b3,
    float* __restrict__ out)
{
  __shared__ float buf[256];
  __shared__ float red[256];
  int g = blockIdx.x, j = threadIdx.x;
  const float* c = comb + (size_t)g * 960;

  float h1 = b1[j];
  for (int k = 0; k < 960; ++k) h1 += c[k] * w1[k * 256 + j];
  float S = blockSum(h1, red);
  float Q = blockSum(h1 * h1, red);
  float mu = S * (1.0f / 256.0f);
  float y  = (h1 - mu) * rsqrtf(fmaxf(Q * (1.0f / 256.0f) - mu * mu, 0.0f) + 1e-5f) * g1[j] + e1[j];
  buf[j] = silu_f(y);
  __syncthreads();

  float h2 = 0.0f;
  if (j < 128) {
    h2 = b2[j];
    for (int k = 0; k < 256; ++k) h2 += buf[k] * w2[k * 128 + j];
  }
  float S2 = blockSum((j < 128) ? h2 : 0.0f, red);
  float Q2 = blockSum((j < 128) ? h2 * h2 : 0.0f, red);
  float contrib = 0.0f;
  if (j < 128) {
    float mu2 = S2 * (1.0f / 128.0f);
    float y2  = (h2 - mu2) * rsqrtf(fmaxf(Q2 * (1.0f / 128.0f) - mu2 * mu2, 0.0f) + 1e-5f) * g2[j] + e2[j];
    contrib = silu_f(y2) * w3[j];
  }
  float S3 = blockSum(contrib, red);
  if (j == 0) out[g] = S3 + b3[0];
}

// ---------------------------------------------------------------------------
extern "C" void kernel_launch(void* const* d_in, const int* in_sizes, int n_in,
                              void* d_out, int out_size, void* d_ws, size_t ws_size,
                              hipStream_t stream) {
  (void)in_sizes; (void)n_in; (void)out_size; (void)ws_size;

  const int*   atoms    = (const int*)  d_in[0];
  const int*   tags     = (const int*)  d_in[1];
  const float* pos      = (const float*)d_in[2];
  const int*   eidx     = (const int*)  d_in[4];
  const int*   batch    = (const int*)  d_in[5];
  const float* atom_emb = (const float*)d_in[6];
  const float* tag_emb  = (const float*)d_in[7];
  const float* pw1      = (const float*)d_in[8];
  const float* pb1      = (const float*)d_in[9];
  const float* pw2      = (const float*)d_in[10];
  const float* pb2      = (const float*)d_in[11];
  const float* in_w     = (const float*)d_in[12];
  const float* in_b     = (const float*)d_in[13];
  float* out = (float*)d_out;

  // ---- workspace carve (256B aligned) ----
  size_t off = 0;
  auto carve = [&](size_t bytes) -> void* {
    void* p = (char*)d_ws + off;
    off += (bytes + 255) & ~(size_t)255;
    return p;
  };
  float*  xf     = (float*) carve((size_t)NNODE * HDIM * 4);       // node state fp32
  float*  agg    = (float*) carve((size_t)NNODE * HDIM * 4);       // neighbor sums (also feat region)
  __bf16* xa     = (__bf16*)carve((size_t)NNODE * HDIM * 2);       // GIN combined (bf16 GEMM A)
  __bf16* hb     = (__bf16*)carve((size_t)NNODE * 2 * HDIM * 2);   // hidden [N,512] bf16
  __bf16* inT    = (__bf16*)carve((size_t)256 * 448 * 2);          // in_w^T
  __bf16* wT     = (__bf16*)carve((size_t)12 * 512 * 256 * 2);     // per layer: w1^T, w2^T
  float*  pooled = (float*) carve((size_t)3 * NGRAPH * 64 * 4);
  float*  addb   = (float*) carve((size_t)NGRAPH * 256 * 4);
  float*  maxb   = (float*) carve((size_t)NGRAPH * 256 * 4);
  float*  counts = (float*) carve((size_t)NGRAPH * 4);
  float*  comb   = (float*) carve((size_t)NGRAPH * 960 * 4);
  __bf16* feat   = (__bf16*)agg;  // [N,448] bf16 lives in agg region pre-layers

  const int NH = NNODE * HDIM;      // 25.6M
  const int mBlocks = NNODE / 32;   // 3125

  // ---- weight conversion (fp32 [K,N] -> bf16 [N,K]) ----
  convert_transpose<<<(448 * 256 + 255) / 256, 256, 0, stream>>>(in_w, inT, 448, 256);
  for (int i = 0; i < 6; ++i) {
    int base = 16 + 10 * i;
    __bf16* w1T = wT + (size_t)i * 2 * 131072;
    __bf16* w2T = w1T + 131072;
    convert_transpose<<<512, 256, 0, stream>>>((const float*)d_in[base + 1], w1T, 256, 512);
    convert_transpose<<<512, 256, 0, stream>>>((const float*)d_in[base + 5], w2T, 512, 256);
  }

  // ---- pooling accumulators ----
  hipMemsetAsync(pooled, 0, (size_t)3 * NGRAPH * 64 * 4, stream);
  hipMemsetAsync(addb, 0, (size_t)NGRAPH * 256 * 4, stream);
  hipMemsetAsync(counts, 0, (size_t)NGRAPH * 4, stream);
  fill_neginf<<<(NGRAPH * 256 + 255) / 256, 256, 0, stream>>>(maxb, NGRAPH * 256);

  // ---- node features + input projection ----
  node_features<<<NNODE, 128, 0, stream>>>(atoms, tags, pos, atom_emb, tag_emb,
                                           pw1, pb1, pw2, pb2, feat);
  gemm_wmma<256, 0><<<mBlocks, 256, 0, stream>>>(feat, inT, in_b, 448,
      nullptr, nullptr, nullptr, nullptr, xf, nullptr);

  // ---- GIN layers ----
  for (int i = 0; i < 6; ++i) {
    int base = 16 + 10 * i;
    const float* epsP   = (const float*)d_in[base + 0];
    const float* b1     = (const float*)d_in[base + 2];
    const float* ln1g   = (const float*)d_in[base + 3];
    const float* ln1b   = (const float*)d_in[base + 4];
    const float* b2     = (const float*)d_in[base + 6];
    const float* ng     = (const float*)d_in[base + 7];
    const float* nb     = (const float*)d_in[base + 8];
    const float* alphaP = (const float*)d_in[base + 9];
    __bf16* w1T = wT + (size_t)i * 2 * 131072;
    __bf16* w2T = w1T + 131072;

    hipMemsetAsync(agg, 0, (size_t)NH * 4, stream);
    scatter_add<<<(NEDGE * 64) / 256, 256, 0, stream>>>(xf, eidx, agg);
    combine_gin<<<NH / 256, 256, 0, stream>>>(xf, agg, epsP, xa);

    gemm_wmma<512, 1><<<mBlocks, 256, 0, stream>>>(xa, w1T, b1, 256,
        ln1g, ln1b, nullptr, nullptr, nullptr, hb);
    gemm_wmma<256, 2><<<mBlocks, 256, 0, stream>>>(hb, w2T, b2, 512,
        ng, nb, alphaP, xf, xf, nullptr);

    if (i == 1 || i == 3 || i == 5) {
      int pi = (i - 1) / 2;
      const float* pw = (const float*)d_in[76 + 2 * pi];
      const float* pb = (const float*)d_in[77 + 2 * pi];
      pool_proj<<<NNODE / 4, 256, 0, stream>>>(xf, pw, pb, batch,
                                               pooled + (size_t)pi * NGRAPH * 64);
    }
  }

  // ---- global pooling + readout ----
  final_pool<<<NH / 256, 256, 0, stream>>>(xf, batch, addb, maxb, counts);
  build_comb<<<(NGRAPH * 960 + 255) / 256, 256, 0, stream>>>(addb, maxb, counts, pooled, comb);
  readout<<<NGRAPH, 256, 0, stream>>>(comb,
      (const float*)d_in[82], (const float*)d_in[83],
      (const float*)d_in[84], (const float*)d_in[85],
      (const float*)d_in[86], (const float*)d_in[87],
      (const float*)d_in[88], (const float*)d_in[89],
      (const float*)d_in[90], (const float*)d_in[91], out);
}
